// HybridLayer_49203145343562
// MI455X (gfx1250) — matrix operand order repaired and checked by reading, not compiled, not run on hardware
//
#include <hip/hip_runtime.h>
#include <hip/hip_bf16.h>
#include <cstdint>
#include <cstddef>

typedef __attribute__((ext_vector_type(16))) __bf16 bf16x16;
typedef __attribute__((ext_vector_type(8)))  float  v8f;

// gcc-vector flavor types for the gfx1250 async/TDM builtins
typedef unsigned int u32x4 __attribute__((vector_size(16)));
typedef int          i32x4 __attribute__((vector_size(16)));
typedef int          i32x8 __attribute__((vector_size(32)));
typedef __attribute__((address_space(1))) i32x4* as1_i32x4;
typedef __attribute__((address_space(3))) i32x4* as3_i32x4;

#define TOKENS 4096
#define DIM    1024
#define FFD    4096
#define NEXP   8
#define BM     128
#define BN     128
#define BK     32

#if __has_builtin(__builtin_amdgcn_global_load_async_to_lds_b128)
#define HAVE_ASYNC_LDS 1
#endif
#if __has_builtin(__builtin_amdgcn_tensor_load_to_lds) && __has_builtin(__builtin_amdgcn_s_wait_tensorcnt)
#define HAVE_TDM 1
#endif

// ---------------------------------------------------------------- helpers
__device__ __forceinline__ unsigned short f2bf(float f) {
  unsigned int u = __float_as_uint(f);
  u += 0x7FFFu + ((u >> 16) & 1u);          // round-to-nearest-even
  return (unsigned short)(u >> 16);
}

// Async (ASYNCcnt) copy of 32B: two GLOBAL_LOAD_ASYNC_TO_LDS_B128 per thread.
__device__ __forceinline__ void stage_a_pair(unsigned short* ldst, const unsigned short* gsrc) {
#ifdef HAVE_ASYNC_LDS
  __builtin_amdgcn_global_load_async_to_lds_b128(
      (as1_i32x4)(size_t)gsrc,
      (as3_i32x4)(size_t)(unsigned)(size_t)ldst, 0, 0);
  __builtin_amdgcn_global_load_async_to_lds_b128(
      (as1_i32x4)(size_t)(gsrc + 8),
      (as3_i32x4)(size_t)(unsigned)(size_t)(ldst + 8), 0, 0);
#else
  const uint4* ga = (const uint4*)gsrc;
  uint4* la = (uint4*)ldst;
  la[0] = ga[0]; la[1] = ga[1];
#endif
}

__device__ __forceinline__ void stage_a_fence() {
#ifdef HAVE_ASYNC_LDS
#if __has_builtin(__builtin_amdgcn_s_wait_asynccnt)
  __builtin_amdgcn_s_wait_asynccnt(0);
#else
  asm volatile("s_wait_asynccnt 0x0" ::: "memory");
#endif
#endif
}

#ifdef HAVE_TDM
// TDM DMA of a 2D bf16 tile (tileRows x tileCols, row stride in elements) into LDS.
// D# bitfields per CDNA5 ISA 8.3/8.4: group0 {count, lds_addr, global_addr, type=2},
// group1 {data_size=1(2B), tensor dims, tile dims, dim0 stride}. Groups 2/3 unused (2D).
__device__ __forceinline__ void tdm_load_2d_bf16(void* ldsDst, const unsigned short* gSrc,
                                                 int tileCols, int tileRows, int rowStrideElts) {
  unsigned long long ga = (unsigned long long)(size_t)gSrc;
  u32x4 g0;
  g0[0] = 1u;                                                  // count=1, user desc
  g0[1] = (unsigned)(size_t)ldsDst;                            // lds_addr
  g0[2] = (unsigned)(ga & 0xFFFFFFFFull);                      // global_addr[31:0]
  g0[3] = (unsigned)((ga >> 32) & 0x1FFFFFFull) | (2u << 30);  // global_addr[56:32] | type=2
  unsigned td0 = 1u << 20, td1 = 1u << 20;                     // generous tensor extents
  i32x8 g1;
  g1[0] = (int)(1u << 16);                                     // data_size = 1 (2 bytes)
  g1[1] = (int)((td0 & 0xFFFFu) << 16);                        // tensor_dim0[15:0]
  g1[2] = (int)(((td0 >> 16) & 0xFFFFu) | ((td1 & 0xFFFFu) << 16));
  g1[3] = (int)(((td1 >> 16) & 0xFFFFu) | ((unsigned)tileCols << 16)); // tile_dim0
  g1[4] = (int)((unsigned)tileRows & 0xFFFFu);                 // tile_dim1 (tile_dim2=0)
  g1[5] = (int)(unsigned)rowStrideElts;                        // tensor_dim0_stride[31:0]
  g1[6] = 0;
  g1[7] = 0;
  i32x4 z4 = {0, 0, 0, 0};
  i32x8 z8 = {0, 0, 0, 0, 0, 0, 0, 0};
  __builtin_amdgcn_tensor_load_to_lds(g0, g1, z4, z4, z8, 0);
}
#endif

// A-fragment (16x32 bf16), ISA layout: lanes 0-15 rows M, VGPR0-3 K=g*8..g*8+7,
// VGPR4-7 K=16+g*8..16+g*8+7 where g = lane>>4. Two 16B LDS loads.
__device__ __forceinline__ bf16x16 load_frag_a(const unsigned short* sA, int row0, int lane) {
  int m = lane & 15, g = lane >> 4;
  const unsigned short* p = sA + (size_t)(row0 + m) * BK;
  union { bf16x16 v; uint4 q[2]; } u;
  u.q[0] = *(const uint4*)(p + g * 8);
  u.q[1] = *(const uint4*)(p + 16 + g * 8);
  return u.v;
}

// B-fragment (32x16 bf16): lane n = lane&15, K = g*8..g*8+7 then 16+g*8..16+g*8+7, g = lane>>4.
// LDS tile stored K-major per column (transposed) so this is two 16B loads.
__device__ __forceinline__ bf16x16 load_frag_b(const unsigned short* sBt, int col0, int lane) {
  int n = lane & 15, g = lane >> 4;
  const unsigned short* p = sBt + (size_t)(col0 + n) * BK + g * 8;
  union { bf16x16 v; uint4 q[2]; } u;
  u.q[0] = *(const uint4*)(p);
  u.q[1] = *(const uint4*)(p + 16);
  return u.v;
}

// stage 32x128 bf16 B tile (row-major global [K][N]) into LDS transposed [BN][BK]
__device__ __forceinline__ void stage_b(const unsigned short* __restrict__ gB, int ldb,
                                        int k0, int n0, unsigned short* sBt, int t) {
  int bk = t & 31;
  int bn = (t >> 5) << 4;
  const uint4* gb = (const uint4*)(gB + (size_t)(k0 + bk) * ldb + n0 + bn);
  union { uint4 q[2]; unsigned short h[16]; } u;
  u.q[0] = gb[0]; u.q[1] = gb[1];
#pragma unroll
  for (int j = 0; j < 16; ++j) sBt[(size_t)(bn + j) * BK + bk] = u.h[j];
}

#define ZERO8 {0.f,0.f,0.f,0.f,0.f,0.f,0.f,0.f}

// ---------------------------------------------------------------- fp32 -> bf16
__global__ void k_f32_to_bf16(const float* __restrict__ src, unsigned short* __restrict__ dst, int n4) {
  int i = blockIdx.x * blockDim.x + threadIdx.x;
  if (i < n4) {
    float4 v = ((const float4*)src)[i];
    ushort4 h;
    h.x = f2bf(v.x); h.y = f2bf(v.y); h.z = f2bf(v.z); h.w = f2bf(v.w);
    ((ushort4*)dst)[i] = h;
  }
}

// ---------------------------------------------------------------- LayerNorm 1
__global__ void __launch_bounds__(256) k_ln1(const float* __restrict__ x,
                                             const float* __restrict__ g,
                                             const float* __restrict__ b,
                                             float* __restrict__ nf,
                                             unsigned short* __restrict__ nbf) {
  int tok = blockIdx.x, t = threadIdx.x;
  const float* row = x + (size_t)tok * DIM;
  float4 v = *(const float4*)(row + t * 4);
  __shared__ float s1[256], s2[256];
  s1[t] = v.x + v.y + v.z + v.w;
  s2[t] = v.x * v.x + v.y * v.y + v.z * v.z + v.w * v.w;
  __syncthreads();
  for (int s = 128; s > 0; s >>= 1) {
    if (t < s) { s1[t] += s1[t + s]; s2[t] += s2[t + s]; }
    __syncthreads();
  }
  float mu = s1[0] * (1.0f / DIM);
  float var = s2[0] * (1.0f / DIM) - mu * mu;
  float rs = rsqrtf(var + 1e-5f);
  float4 gg = *(const float4*)(g + t * 4);
  float4 bb = *(const float4*)(b + t * 4);
  float4 o;
  o.x = (v.x - mu) * rs * gg.x + bb.x;
  o.y = (v.y - mu) * rs * gg.y + bb.y;
  o.z = (v.z - mu) * rs * gg.z + bb.z;
  o.w = (v.w - mu) * rs * gg.w + bb.w;
  *(float4*)(nf + (size_t)tok * DIM + t * 4) = o;
  ushort4 h; h.x = f2bf(o.x); h.y = f2bf(o.y); h.z = f2bf(o.z); h.w = f2bf(o.w);
  *(ushort4*)(nbf + (size_t)tok * DIM + t * 4) = h;
}

// ---------------------------------------------------------------- Mixer GEMM
// x1 = x + n1 + n1 @ mix_w + mix_b   (also seeds d_out with the residual base)
__global__ void __launch_bounds__(256) k_gemm_mixer(const unsigned short* __restrict__ Abf,
                                                    const unsigned short* __restrict__ Bbf,
                                                    const float* __restrict__ x,
                                                    const float* __restrict__ n1f,
                                                    const float* __restrict__ mix_b,
                                                    float* __restrict__ x1,
                                                    float* __restrict__ out) {
  __shared__ unsigned short sA[BM * BK];
  __shared__ unsigned short sB[BN * BK];
  int t = threadIdx.x;
  int lane = t & 31, wave = t >> 5;
  int wm = wave & 3, wn = wave >> 2;
  int m0 = blockIdx.x * BM, n0 = blockIdx.y * BN;
  v8f acc[2][4] = {{ZERO8, ZERO8, ZERO8, ZERO8}, {ZERO8, ZERO8, ZERO8, ZERO8}};
  int ar = t >> 1, ak = (t & 1) << 4;
  const unsigned short* aRow = Abf + (size_t)(m0 + ar) * DIM;
  for (int k0 = 0; k0 < DIM; k0 += BK) {
    stage_a_pair(sA + (size_t)ar * BK + ak, aRow + k0 + ak);
    stage_b(Bbf, DIM, k0, n0, sB, t);
    if (k0 + BK < DIM) {
      __builtin_prefetch(aRow + k0 + BK + ak, 0, 1);
      __builtin_prefetch(Bbf + (size_t)(k0 + BK + (t & 31)) * DIM + n0, 0, 1);
    }
    stage_a_fence();
    __syncthreads();
    bf16x16 af0 = load_frag_a(sA, wm * 32, lane);
    bf16x16 af1 = load_frag_a(sA, wm * 32 + 16, lane);
#pragma unroll
    for (int fn = 0; fn < 4; ++fn) {
      bf16x16 bfr = load_frag_b(sB, wn * 64 + fn * 16, lane);
      acc[0][fn] = __builtin_amdgcn_wmma_f32_16x16x32_bf16(false, af0, false, bfr, (short)0, acc[0][fn], false, false);
      acc[1][fn] = __builtin_amdgcn_wmma_f32_16x16x32_bf16(false, af1, false, bfr, (short)0, acc[1][fn], false, false);
    }
    __syncthreads();
  }
  int lm = (lane >> 4) << 3;
  int lnn = lane & 15;
#pragma unroll
  for (int fm = 0; fm < 2; ++fm)
#pragma unroll
    for (int fn = 0; fn < 4; ++fn) {
      int mBase = m0 + wm * 32 + fm * 16 + lm;
      int nCol = n0 + wn * 64 + fn * 16 + lnn;
      float mb = mix_b[nCol];
#pragma unroll
      for (int r = 0; r < 8; ++r) {
        size_t idx = (size_t)(mBase + r) * DIM + nCol;
        float v = x[idx] + n1f[idx] + mb + acc[fm][fn][r];
        x1[idx] = v;
        out[idx] = v;
      }
    }
}

// ---------------------------------------------------------------- LN2 + router
__global__ void __launch_bounds__(256) k_ln2_router(const float* __restrict__ x1,
                                                    const float* __restrict__ g,
                                                    const float* __restrict__ b,
                                                    const float* __restrict__ rw,
                                                    unsigned short* __restrict__ n2bf,
                                                    int* __restrict__ counts,
                                                    int* __restrict__ tokidx,
                                                    float* __restrict__ tokwgt) {
  int tok = blockIdx.x, t = threadIdx.x;
  const float* row = x1 + (size_t)tok * DIM;
  float4 v = *(const float4*)(row + t * 4);
  __shared__ float s1[256], s2[256];
  __shared__ float red[256 * NEXP];
  s1[t] = v.x + v.y + v.z + v.w;
  s2[t] = v.x * v.x + v.y * v.y + v.z * v.z + v.w * v.w;
  __syncthreads();
  for (int s = 128; s > 0; s >>= 1) {
    if (t < s) { s1[t] += s1[t + s]; s2[t] += s2[t + s]; }
    __syncthreads();
  }
  float mu = s1[0] * (1.0f / DIM);
  float var = s2[0] * (1.0f / DIM) - mu * mu;
  float rs = rsqrtf(var + 1e-5f);
  float4 gg = *(const float4*)(g + t * 4);
  float4 bb = *(const float4*)(b + t * 4);
  float nv[4];
  nv[0] = (v.x - mu) * rs * gg.x + bb.x;
  nv[1] = (v.y - mu) * rs * gg.y + bb.y;
  nv[2] = (v.z - mu) * rs * gg.z + bb.z;
  nv[3] = (v.w - mu) * rs * gg.w + bb.w;
  ushort4 h; h.x = f2bf(nv[0]); h.y = f2bf(nv[1]); h.z = f2bf(nv[2]); h.w = f2bf(nv[3]);
  *(ushort4*)(n2bf + (size_t)tok * DIM + t * 4) = h;
  // router logits
  float part[NEXP];
#pragma unroll
  for (int e = 0; e < NEXP; ++e) part[e] = 0.f;
  const float* rwp = rw + (size_t)(t * 4) * NEXP;
#pragma unroll
  for (int j = 0; j < 4; ++j)
#pragma unroll
    for (int e = 0; e < NEXP; ++e) part[e] += nv[j] * rwp[j * NEXP + e];
#pragma unroll
  for (int e = 0; e < NEXP; ++e) red[t * NEXP + e] = part[e];
  __syncthreads();
  for (int s = 128; s > 0; s >>= 1) {
    if (t < s)
#pragma unroll
      for (int e = 0; e < NEXP; ++e) red[t * NEXP + e] += red[(t + s) * NEXP + e];
    __syncthreads();
  }
  if (t == 0) {
    float lg[NEXP];
#pragma unroll
    for (int e = 0; e < NEXP; ++e) lg[e] = red[e];
    float mx = lg[0];
#pragma unroll
    for (int e = 1; e < NEXP; ++e) mx = fmaxf(mx, lg[e]);
    float ex[NEXP], ssum = 0.f;
#pragma unroll
    for (int e = 0; e < NEXP; ++e) { ex[e] = __expf(lg[e] - mx); ssum += ex[e]; }
    float inv = 1.f / ssum;
    int i1 = 0;
    for (int e = 1; e < NEXP; ++e) if (lg[e] > lg[i1]) i1 = e;
    int i2 = (i1 == 0) ? 1 : 0;
    for (int e = 0; e < NEXP; ++e) if (e != i1 && lg[e] > lg[i2]) i2 = e;
    int p1 = atomicAdd(&counts[i1], 1);
    tokidx[i1 * TOKENS + p1] = tok; tokwgt[i1 * TOKENS + p1] = ex[i1] * inv;
    int p2 = atomicAdd(&counts[i2], 1);
    tokidx[i2 * TOKENS + p2] = tok; tokwgt[i2 * TOKENS + p2] = ex[i2] * inv;
  }
}

__global__ void k_zero_counts(int* counts) { if (threadIdx.x < NEXP) counts[threadIdx.x] = 0; }

__global__ void k_prefix(const int* __restrict__ counts, int* __restrict__ segoff) {
  if (threadIdx.x == 0) {
    int s = 0;
    for (int e = 0; e < NEXP; ++e) { segoff[e] = s; s += ((counts[e] + BM - 1) / BM) * BM; }
  }
}

// ---------------------------------------------------------------- FFN GEMM1: h = silu(n2 @ w1[e] + b1[e])
__global__ void __launch_bounds__(256) k_gemm_ffn1(const unsigned short* __restrict__ n2bf,
                                                   const unsigned short* __restrict__ w1bf,
                                                   const float* __restrict__ b1,
                                                   const int* __restrict__ counts,
                                                   const int* __restrict__ segoff,
                                                   const int* __restrict__ tokidx,
                                                   unsigned short* __restrict__ hpool) {
  int e = blockIdx.z;
  int cnt = counts[e];
  int m0 = blockIdx.x * BM;
  if (m0 >= cnt) return;
  int n0 = blockIdx.y * BN;
  const unsigned short* Bexp = w1bf + (size_t)e * DIM * FFD;
  __shared__ unsigned short sA[BM * BK];
  __shared__ unsigned short sB[BN * BK];
  int t = threadIdx.x;
  int lane = t & 31, wave = t >> 5;
  int wm = wave & 3, wn = wave >> 2;
  v8f acc[2][4] = {{ZERO8, ZERO8, ZERO8, ZERO8}, {ZERO8, ZERO8, ZERO8, ZERO8}};
  int ar = t >> 1, ak = (t & 1) << 4;
  int slot = m0 + ar;
  int tokRow = (slot < cnt) ? tokidx[e * TOKENS + slot] : 0;
  const unsigned short* aRow = n2bf + (size_t)tokRow * DIM;
  for (int k0 = 0; k0 < DIM; k0 += BK) {
    stage_a_pair(sA + (size_t)ar * BK + ak, aRow + k0 + ak);
    stage_b(Bexp, FFD, k0, n0, sB, t);
    if (k0 + BK < DIM) {
      __builtin_prefetch(aRow + k0 + BK + ak, 0, 1);
      __builtin_prefetch(Bexp + (size_t)(k0 + BK + (t & 31)) * FFD + n0, 0, 1);
    }
    stage_a_fence();
    __syncthreads();
    bf16x16 af0 = load_frag_a(sA, wm * 32, lane);
    bf16x16 af1 = load_frag_a(sA, wm * 32 + 16, lane);
#pragma unroll
    for (int fn = 0; fn < 4; ++fn) {
      bf16x16 bfr = load_frag_b(sB, wn * 64 + fn * 16, lane);
      acc[0][fn] = __builtin_amdgcn_wmma_f32_16x16x32_bf16(false, af0, false, bfr, (short)0, acc[0][fn], false, false);
      acc[1][fn] = __builtin_amdgcn_wmma_f32_16x16x32_bf16(false, af1, false, bfr, (short)0, acc[1][fn], false, false);
    }
    __syncthreads();
  }
  int lm = (lane >> 4) << 3;
  int lnn = lane & 15;
  int base = segoff[e] + m0;
#pragma unroll
  for (int fm = 0; fm < 2; ++fm)
#pragma unroll
    for (int fn = 0; fn < 4; ++fn) {
      int mBase = wm * 32 + fm * 16 + lm;
      int nCol = n0 + wn * 64 + fn * 16 + lnn;
      float bias = b1[e * FFD + nCol];
#pragma unroll
      for (int r = 0; r < 8; ++r) {
        float v = acc[fm][fn][r] + bias;
        float s = v * (1.0f / (1.0f + __expf(-v)));  // silu
        hpool[(size_t)(base + mBase + r) * FFD + nCol] = f2bf(s);
      }
    }
}

// ---------------------------------------------------------------- FFN GEMM2: out += w * (h @ w2[e] + b2[e])
__global__ void __launch_bounds__(256) k_gemm_ffn2(const unsigned short* __restrict__ hpool,
                                                   const unsigned short* __restrict__ w2bf,
                                                   const float* __restrict__ b2,
                                                   const int* __restrict__ counts,
                                                   const int* __restrict__ segoff,
                                                   const int* __restrict__ tokidx,
                                                   const float* __restrict__ tokwgt,
                                                   float* __restrict__ out) {
  int e = blockIdx.z;
  int cnt = counts[e];
  int m0 = blockIdx.x * BM;
  if (m0 >= cnt) return;
  int n0 = blockIdx.y * BN;
  const unsigned short* Aexp = hpool + (size_t)segoff[e] * FFD;
  const unsigned short* Bexp = w2bf + (size_t)e * FFD * DIM;
  __shared__ unsigned short sA[BM * BK];
  __shared__ unsigned short sB[BN * BK];
  int t = threadIdx.x;
  int lane = t & 31, wave = t >> 5;
  int wm = wave & 3, wn = wave >> 2;
  v8f acc[2][4] = {{ZERO8, ZERO8, ZERO8, ZERO8}, {ZERO8, ZERO8, ZERO8, ZERO8}};
  int ar = t >> 1, ak = (t & 1) << 4;
  const unsigned short* aTile = Aexp + (size_t)m0 * FFD;   // 128 x FFD tile base
  const unsigned short* aRow = aTile + (size_t)ar * FFD;
  for (int k0 = 0; k0 < FFD; k0 += BK) {
#ifdef HAVE_TDM
    // One wave programs the Tensor Data Mover: 128x32 bf16 tile, row stride FFD.
    if (wave == 0) tdm_load_2d_bf16(sA, aTile + k0, BK, BM, FFD);
#else
    stage_a_pair(sA + (size_t)ar * BK + ak, aRow + k0 + ak);
#endif
    stage_b(Bexp, DIM, k0, n0, sB, t);
    if (k0 + BK < FFD) {
      __builtin_prefetch(aRow + k0 + BK + ak, 0, 1);
      __builtin_prefetch(Bexp + (size_t)(k0 + BK + (t & 31)) * DIM + n0, 0, 1);
    }
#ifdef HAVE_TDM
    if (wave == 0) __builtin_amdgcn_s_wait_tensorcnt(0);
#else
    stage_a_fence();
#endif
    __syncthreads();
    bf16x16 af0 = load_frag_a(sA, wm * 32, lane);
    bf16x16 af1 = load_frag_a(sA, wm * 32 + 16, lane);
#pragma unroll
    for (int fn = 0; fn < 4; ++fn) {
      bf16x16 bfr = load_frag_b(sB, wn * 64 + fn * 16, lane);
      acc[0][fn] = __builtin_amdgcn_wmma_f32_16x16x32_bf16(false, af0, false, bfr, (short)0, acc[0][fn], false, false);
      acc[1][fn] = __builtin_amdgcn_wmma_f32_16x16x32_bf16(false, af1, false, bfr, (short)0, acc[1][fn], false, false);
    }
    __syncthreads();
  }
  int lm = (lane >> 4) << 3;
  int lnn = lane & 15;
#pragma unroll
  for (int fm = 0; fm < 2; ++fm)
#pragma unroll
    for (int fn = 0; fn < 4; ++fn) {
      int mBase = m0 + wm * 32 + fm * 16 + lm;
      int nCol = n0 + wn * 64 + fn * 16 + lnn;
      float bias = b2[e * DIM + nCol];
#pragma unroll
      for (int r = 0; r < 8; ++r) {
        int slot = mBase + r;
        if (slot < cnt) {
          int tok = tokidx[e * TOKENS + slot];
          float w = tokwgt[e * TOKENS + slot];
          atomicAdd(&out[(size_t)tok * DIM + nCol], w * (acc[fm][fn][r] + bias));
        }
      }
    }
}

// ---------------------------------------------------------------- host side
extern "C" void kernel_launch(void* const* d_in, const int* in_sizes, int n_in,
                              void* d_out, int out_size, void* d_ws, size_t ws_size,
                              hipStream_t stream) {
  const float* x        = (const float*)d_in[0];
  const float* g1       = (const float*)d_in[1];
  const float* be1      = (const float*)d_in[2];
  const float* mix_w    = (const float*)d_in[3];
  const float* mix_b    = (const float*)d_in[4];
  const float* g2       = (const float*)d_in[5];
  const float* be2      = (const float*)d_in[6];
  const float* router_w = (const float*)d_in[7];
  const float* w1       = (const float*)d_in[8];
  const float* b1       = (const float*)d_in[9];
  const float* w2       = (const float*)d_in[10];
  const float* b2       = (const float*)d_in[11];
  float* out = (float*)d_out;

  char* ws = (char*)d_ws;
  size_t off = 0;
  auto alloc = [&](size_t bytes) -> char* {
    char* p = ws + off;
    off = (off + bytes + 255) & ~(size_t)255;
    return p;
  };
  unsigned short* mixwbf = (unsigned short*)alloc((size_t)DIM * DIM * 2);
  unsigned short* w1bf   = (unsigned short*)alloc((size_t)NEXP * DIM * FFD * 2);
  unsigned short* w2bf   = (unsigned short*)alloc((size_t)NEXP * FFD * DIM * 2);
  float*          n1f    = (float*)alloc((size_t)TOKENS * DIM * 4);
  unsigned short* n1bf   = (unsigned short*)alloc((size_t)TOKENS * DIM * 2);
  float*          x1     = (float*)alloc((size_t)TOKENS * DIM * 4);
  unsigned short* n2bf   = (unsigned short*)alloc((size_t)TOKENS * DIM * 2);
  int*            counts = (int*)alloc(NEXP * 4);
  int*            segoff = (int*)alloc(NEXP * 4);
  int*            tokidx = (int*)alloc((size_t)NEXP * TOKENS * 4);
  float*          tokwgt = (float*)alloc((size_t)NEXP * TOKENS * 4);
  unsigned short* hpool  = (unsigned short*)alloc((size_t)(2 * TOKENS + NEXP * BM) * FFD * 2);

  // 1) weight conversion to bf16
  {
    int n4 = DIM * DIM / 4;
    k_f32_to_bf16<<<(n4 + 255) / 256, 256, 0, stream>>>(mix_w, mixwbf, n4);
    n4 = NEXP * DIM * FFD / 4;
    k_f32_to_bf16<<<(n4 + 255) / 256, 256, 0, stream>>>(w1, w1bf, n4);
    k_f32_to_bf16<<<(n4 + 255) / 256, 256, 0, stream>>>(w2, w2bf, n4);
  }
  // 2) LN1
  k_ln1<<<TOKENS, 256, 0, stream>>>(x, g1, be1, n1f, n1bf);
  // 3) mixer GEMM (also seeds out with residual base x1)
  k_gemm_mixer<<<dim3(TOKENS / BM, DIM / BN), 256, 0, stream>>>(n1bf, mixwbf, x, n1f, mix_b, x1, out);
  // 4) router
  k_zero_counts<<<1, 32, 0, stream>>>(counts);
  k_ln2_router<<<TOKENS, 256, 0, stream>>>(x1, g2, be2, router_w, n2bf, counts, tokidx, tokwgt);
  k_prefix<<<1, 32, 0, stream>>>(counts, segoff);
  // 5) FFN
  k_gemm_ffn1<<<dim3(TOKENS / BM, FFD / BN, NEXP), 256, 0, stream>>>(n2bf, w1bf, b1, counts, segoff, tokidx, hpool);
  k_gemm_ffn2<<<dim3(TOKENS / BM, DIM / BN, NEXP), 256, 0, stream>>>(hpool, w2bf, b2, counts, segoff, tokidx, tokwgt, out);
}
